// HunyuanVideoDualAttention_5437428596770
// MI455X (gfx1250) — matrix operand order repaired and checked by reading, not compiled, not run on hardware
//
#include <hip/hip_runtime.h>
#include <hip/hip_bf16.h>

typedef __bf16 bf16;
typedef __attribute__((ext_vector_type(16))) __bf16 v16bf;
typedef __attribute__((ext_vector_type(8)))  __bf16 v8bf;
typedef __attribute__((ext_vector_type(8)))  float  v8f;

union V16U { v16bf v; v8bf h[2]; };
struct alignas(8) BF4 { bf16 x, y, z, w; };

__device__ __forceinline__ bf16 f2bf(float f) {
  unsigned u = __float_as_uint(f);
  unsigned r = (u + 0x7FFFu + ((u >> 16) & 1u)) >> 16;
  unsigned short s = (unsigned short)r;
  return __builtin_bit_cast(bf16, s);
}

// Async 16B global -> LDS copy (ASYNCcnt-tracked). LDS operand is the 32-bit
// LDS byte offset (low 32 bits of the generic shared-memory address, per the
// CDNA5 flat-aperture truncation rule).
__device__ __forceinline__ void async_b128(void* lds_ptr, const void* gptr) {
  unsigned lo = (unsigned)(unsigned long long)(size_t)lds_ptr;
  unsigned long long ga = (unsigned long long)(size_t)gptr;
  asm volatile("global_load_async_to_lds_b128 %0, %1, off"
               :: "v"(lo), "v"(ga) : "memory");
}
__device__ __forceinline__ void wait_async0() {
  asm volatile("s_wait_asynccnt 0" ::: "memory");
}

// ---------------------------------------------------------------------------
// Elementwise fp32 -> bf16 (weights + activations preconversion)
// ---------------------------------------------------------------------------
__global__ void convert_f32_bf16(const float* __restrict__ src,
                                 bf16* __restrict__ dst, int n4) {
  int i = blockIdx.x * blockDim.x + threadIdx.x;
  if (i >= n4) return;
  float4 v = ((const float4*)src)[i];
  BF4 o{f2bf(v.x), f2bf(v.y), f2bf(v.z), f2bf(v.w)};
  ((BF4*)dst)[i] = o;
}

// ---------------------------------------------------------------------------
// bf16 WMMA GEMM:  C[M][N] (f32) = A[M][K] @ B[N][K]^T + bias[N]
// Block tile 128x128x32, 256 threads = 8 waves in 2x4; wave tile 64x32.
// Double-buffered LDS filled with global_load_async_to_lds_b128.
// ---------------------------------------------------------------------------
__global__ __launch_bounds__(256) void gemm_wmma_bf16(
    const bf16* __restrict__ A, const bf16* __restrict__ B,
    const float* __restrict__ bias, float* __restrict__ C,
    int M, int N, int K)
{
  constexpr int BM = 128, BN = 128, BK = 32;
  constexpr int LDT = BK + 8;              // padded LDS row stride (elems)
  __shared__ bf16 As[2][BM * LDT];
  __shared__ bf16 Bs[2][BN * LDT];

  const int t     = threadIdx.x;
  const int lane  = t & 31;
  const int wid   = t >> 5;
  const int waveM = wid >> 2;              // 0..1
  const int waveN = wid & 3;               // 0..3
  const int l16   = lane & 15;
  const int lhi   = lane >> 4;             // 0/1

  const int m0 = blockIdx.y * BM;
  const int n0 = blockIdx.x * BN;

  v8f acc[4][2] = {};

  auto issue_tile = [&](int buf, int k0) {
    #pragma unroll
    for (int it = 0; it < 2; ++it) {
      int idx = t + 256 * it;              // 512 slots x 8 bf16 per matrix
      int row = idx >> 2, c8 = idx & 3;
      async_b128(&As[buf][row * LDT + c8 * 8],
                 A + (size_t)(m0 + row) * K + k0 + c8 * 8);
      async_b128(&Bs[buf][row * LDT + c8 * 8],
                 B + (size_t)(n0 + row) * K + k0 + c8 * 8);
    }
  };

  issue_tile(0, 0);
  wait_async0();
  __syncthreads();

  int buf = 0;
  for (int k0 = 0; k0 < K; k0 += BK) {
    if (k0 + BK < K) issue_tile(buf ^ 1, k0 + BK);

    const bf16* as = As[buf];
    const bf16* bs = Bs[buf];
    V16U bfrag[2];
    #pragma unroll
    for (int j = 0; j < 2; ++j) {
      const bf16* bp = bs + (waveN * 32 + j * 16 + l16) * LDT + lhi * 8;
      bfrag[j].h[0] = *(const v8bf*)bp;
      bfrag[j].h[1] = *(const v8bf*)(bp + 16);
    }
    #pragma unroll
    for (int i = 0; i < 4; ++i) {
      V16U afrag;
      const bf16* ap = as + (waveM * 64 + i * 16 + l16) * LDT + lhi * 8;
      afrag.h[0] = *(const v8bf*)ap;
      afrag.h[1] = *(const v8bf*)(ap + 16);
      #pragma unroll
      for (int j = 0; j < 2; ++j)
        acc[i][j] = __builtin_amdgcn_wmma_f32_16x16x32_bf16(
            false, afrag.v, false, bfrag[j].v, (short)0, acc[i][j], false, false);
    }

    wait_async0();
    __syncthreads();
    buf ^= 1;
  }

  // ---- epilogue: bias + f32 store ----
  #pragma unroll
  for (int j = 0; j < 2; ++j) {
    int col = n0 + waveN * 32 + j * 16 + l16;
    float bv = bias ? bias[col] : 0.f;
    #pragma unroll
    for (int i = 0; i < 4; ++i) {
      int rbase = m0 + waveM * 64 + i * 16 + lhi * 8;
      #pragma unroll
      for (int r = 0; r < 8; ++r)
        C[(size_t)(rbase + r) * N + col] = acc[i][j][r] + bv;
    }
  }
}

// ---------------------------------------------------------------------------
// RMSNorm + interleaved RoPE + 1/sqrt(D) fold + pack Q/K [H][S][D], V^T [H][D][S]
// ---------------------------------------------------------------------------
__global__ __launch_bounds__(128) void qkv_prep(
    const float* __restrict__ qkv,   // [2048][9216]
    const float* __restrict__ eqkv,  // [256][9216]
    const float* __restrict__ cosb, const float* __restrict__ sinb, // [2048][64]
    const float* __restrict__ nq,  const float* __restrict__ nk,
    const float* __restrict__ naq, const float* __restrict__ nak,
    bf16* __restrict__ Qb, bf16* __restrict__ Kb, bf16* __restrict__ Vt)
{
  constexpr int S = 2304, D = 128;
  const int s = blockIdx.x;
  const int d = threadIdx.x;
  const bool img = (s < 2048);
  const float* src = img ? (qkv + (size_t)s * 9216)
                         : (eqkv + (size_t)(s - 2048) * 9216);
  const float* wq = img ? nq : naq;
  const float* wk = img ? nk : nak;
  float c = 1.f, sn = 0.f;
  if (img) { c = cosb[s * 64 + (d >> 1)]; sn = sinb[s * 64 + (d >> 1)]; }

  __shared__ float part[4];
  const int lane = d & 31, wv = d >> 5;
  const float qscale = 0.08838834764831845f;   // 1/sqrt(128)

  for (int h = 0; h < 24; ++h) {
    {   // ---- Q: rmsnorm + rope + scale ----
      const float* x = src + h * D;
      float v = x[d];
      float ss = v * v;
      #pragma unroll
      for (int m = 16; m >= 1; m >>= 1) ss += __shfl_xor(ss, m, 32);
      if (lane == 0) part[wv] = ss;
      __syncthreads();
      float tot = part[0] + part[1] + part[2] + part[3];
      __syncthreads();
      float rstd = rsqrtf(tot * (1.f / 128.f) + 1e-6f);
      float xn = v * rstd * wq[d] * qscale;
      float out = xn;
      if (img) {
        float pv = x[d ^ 1] * rstd * wq[d ^ 1] * qscale;
        out = (d & 1) ? (xn * c + pv * sn) : (xn * c - pv * sn);
      }
      Qb[((size_t)h * S + s) * D + d] = f2bf(out);
    }
    {   // ---- K: rmsnorm + rope ----
      const float* x = src + 3072 + h * D;
      float v = x[d];
      float ss = v * v;
      #pragma unroll
      for (int m = 16; m >= 1; m >>= 1) ss += __shfl_xor(ss, m, 32);
      if (lane == 0) part[wv] = ss;
      __syncthreads();
      float tot = part[0] + part[1] + part[2] + part[3];
      __syncthreads();
      float rstd = rsqrtf(tot * (1.f / 128.f) + 1e-6f);
      float xn = v * rstd * wk[d];
      float out = xn;
      if (img) {
        float pv = x[d ^ 1] * rstd * wk[d ^ 1];
        out = (d & 1) ? (xn * c + pv * sn) : (xn * c - pv * sn);
      }
      Kb[((size_t)h * S + s) * D + d] = f2bf(out);
    }
    // ---- V: transpose pack ----
    Vt[((size_t)h * D + d) * S + s] = f2bf(src[6144 + h * D + d]);
  }
}

// ---------------------------------------------------------------------------
// Flash attention, bf16 WMMA. RMS-normed Q/K bound |score| <= sqrt(128), so
// exp() cannot overflow fp32 -> no running max needed. Row sums are computed
// with a ones-matrix WMMA whose C-layout aligns row-for-row with the output
// accumulator (no cross-lane shuffles at all).
// grid (36, 24): 64 queries x head per block; 4 waves, 16 queries per wave.
// K/V tiles double-buffered via global_load_async_to_lds_b128.
// ---------------------------------------------------------------------------
__global__ __launch_bounds__(128) void attn_wmma(
    const bf16* __restrict__ Qb, const bf16* __restrict__ Kb,
    const bf16* __restrict__ Vt, bf16* __restrict__ O /* [2304][3072] */)
{
  constexpr int S = 2304, D = 128;
  constexpr int LK = D + 8;    // Ks row stride
  constexpr int LV = 40;       // Vs row stride
  constexpr int LP = 40;       // Ps row stride
  __shared__ bf16 Ks[2][32 * LK];      // [key][d]
  __shared__ bf16 Vs[2][128 * LV];     // [d][key]
  __shared__ bf16 Ps[4][16 * LP];      // per-wave P tile [q][key]

  const int h    = blockIdx.y;
  const int t    = threadIdx.x;
  const int lane = t & 31;
  const int wid  = t >> 5;
  const int l16  = lane & 15, lhi = lane >> 4;
  const int qbase = blockIdx.x * 64 + wid * 16;

  // Q fragments (A layout), kept in registers for all key chunks
  V16U qf[4];
  {
    const bf16* qp = Qb + ((size_t)h * S + qbase + l16) * D;
    #pragma unroll
    for (int ks = 0; ks < 4; ++ks) {
      const bf16* p = qp + ks * 32 + lhi * 8;
      qf[ks].h[0] = *(const v8bf*)p;
      qf[ks].h[1] = *(const v8bf*)(p + 16);
    }
  }

  // all-ones B fragment for row-sum WMMA
  V16U ones;
  #pragma unroll
  for (int e = 0; e < 8; ++e) { ones.h[0][e] = f2bf(1.f); ones.h[1][e] = f2bf(1.f); }

  v8f oacc[8] = {};
  v8f sumacc = {};

  auto issue_kv = [&](int buf, int k0) {
    #pragma unroll
    for (int it = 0; it < 4; ++it) {
      int idx = t + 128 * it;                 // 512 slots of 8 bf16
      int kr = idx >> 4, kc = idx & 15;       // key, d-chunk
      async_b128(&Ks[buf][kr * LK + kc * 8],
                 Kb + ((size_t)h * S + k0 + kr) * D + kc * 8);
      int vr = idx >> 2, vc = idx & 3;        // d, key-chunk
      async_b128(&Vs[buf][vr * LV + vc * 8],
                 Vt + ((size_t)h * D + vr) * S + k0 + vc * 8);
    }
  };

  issue_kv(0, 0);
  wait_async0();
  __syncthreads();

  int buf = 0;
  for (int k0 = 0; k0 < S; k0 += 32) {
    if (k0 + 32 < S) issue_kv(buf ^ 1, k0 + 32);

    const bf16* ks_ = Ks[buf];
    const bf16* vs_ = Vs[buf];

    // ---- scores: Q (16xD) @ K^T (Dx32) -> two 16x16 f32 tiles ----
    v8f sacc[2] = {};
    #pragma unroll
    for (int tt = 0; tt < 2; ++tt) {
      #pragma unroll
      for (int kk = 0; kk < 4; ++kk) {
        V16U bfr;
        const bf16* bp = ks_ + (tt * 16 + l16) * LK + kk * 32 + lhi * 8;
        bfr.h[0] = *(const v8bf*)bp;
        bfr.h[1] = *(const v8bf*)(bp + 16);
        sacc[tt] = __builtin_amdgcn_wmma_f32_16x16x32_bf16(
            false, qf[kk].v, false, bfr.v, (short)0, sacc[tt], false, false);
      }
    }

    // ---- P = exp(score); re-swizzle C-layout -> A-layout through LDS ----
    #pragma unroll
    for (int r = 0; r < 8; ++r) {
      bf16* pp = Ps[wid] + (r + 8 * lhi) * LP;
      pp[l16]      = f2bf(__expf(sacc[0][r]));
      pp[16 + l16] = f2bf(__expf(sacc[1][r]));
    }
    V16U pf;
    {
      const bf16* p = Ps[wid] + l16 * LP + lhi * 8;
      pf.h[0] = *(const v8bf*)p;
      pf.h[1] = *(const v8bf*)(p + 16);
    }

    // ---- row sums via ones-WMMA (aligned with oacc rows) ----
    sumacc = __builtin_amdgcn_wmma_f32_16x16x32_bf16(
        false, pf.v, false, ones.v, (short)0, sumacc, false, false);

    // ---- O += P (16x32) @ V (32xD) ----
    #pragma unroll
    for (int n = 0; n < 8; ++n) {
      V16U bfr;
      const bf16* bp = vs_ + (n * 16 + l16) * LV + lhi * 8;
      bfr.h[0] = *(const v8bf*)bp;
      bfr.h[1] = *(const v8bf*)(bp + 16);
      oacc[n] = __builtin_amdgcn_wmma_f32_16x16x32_bf16(
          false, pf.v, false, bfr.v, (short)0, oacc[n], false, false);
    }

    wait_async0();
    __syncthreads();
    buf ^= 1;
  }

  // ---- normalize and store attn output (bf16, [S][DM]) ----
  #pragma unroll
  for (int r = 0; r < 8; ++r) {
    float inv = 1.f / sumacc[r];
    int row = qbase + r + 8 * lhi;
    #pragma unroll
    for (int n = 0; n < 8; ++n)
      O[(size_t)row * 3072 + h * 128 + n * 16 + l16] = f2bf(oacc[n][r] * inv);
  }
}

// ---------------------------------------------------------------------------
// Host launcher
// ---------------------------------------------------------------------------
extern "C" void kernel_launch(void* const* d_in, const int* in_sizes, int n_in,
                              void* d_out, int out_size, void* d_ws, size_t ws_size,
                              hipStream_t stream)
{
  (void)in_sizes; (void)n_in; (void)out_size; (void)ws_size;
  const float* hidden = (const float*)d_in[0];
  const float* enc    = (const float*)d_in[1];
  const float* cosb   = (const float*)d_in[2];
  const float* sinb   = (const float*)d_in[3];
  const float* w_qkv  = (const float*)d_in[4];
  const float* b_qkv  = (const float*)d_in[5];
  const float* w_aqkv = (const float*)d_in[6];
  const float* b_aqkv = (const float*)d_in[7];
  const float* w_out  = (const float*)d_in[8];
  const float* b_out  = (const float*)d_in[9];
  const float* w_aout = (const float*)d_in[10];
  const float* b_aout = (const float*)d_in[11];
  const float* nq  = (const float*)d_in[12];
  const float* nk  = (const float*)d_in[13];
  const float* naq = (const float*)d_in[14];
  const float* nak = (const float*)d_in[15];
  float* out = (float*)d_out;

  char* ws = (char*)d_ws;
  size_t off = 0;
  bf16*  wqkv_bf  = (bf16*)(ws + off);  off += 56623104;   // 9216x3072 bf16
  bf16*  waqkv_bf = (bf16*)(ws + off);  off += 56623104;
  bf16*  wout_bf  = (bf16*)(ws + off);  off += 18874368;   // 3072x3072 bf16
  bf16*  waout_bf = (bf16*)(ws + off);  off += 18874368;
  bf16*  hid_bf   = (bf16*)(ws + off);  off += 12582912;   // 2048x3072 bf16
  bf16*  enc_bf   = (bf16*)(ws + off);  off += 1572864;    // 256x3072 bf16
  float* qkv      = (float*)(ws + off); off += 75497472;   // 2048x9216 f32
  float* eqkv     = (float*)(ws + off); off += 9437184;    // 256x9216 f32
  bf16*  Qb       = (bf16*)(ws + off);  off += 14155776;   // 24x2304x128 bf16
  bf16*  Kb       = (bf16*)(ws + off);  off += 14155776;
  bf16*  Vt       = (bf16*)(ws + off);  off += 14155776;
  bf16*  attn     = (bf16*)(ws + off);  off += 14155776;   // 2304x3072 bf16

  auto cvt = [&](const float* s, bf16* d, size_t n) {
    int n4 = (int)(n / 4);
    convert_f32_bf16<<<(n4 + 255) / 256, 256, 0, stream>>>(s, d, n4);
  };
  cvt(w_qkv,  wqkv_bf,  9216ull * 3072);
  cvt(w_aqkv, waqkv_bf, 9216ull * 3072);
  cvt(w_out,  wout_bf,  3072ull * 3072);
  cvt(w_aout, waout_bf, 3072ull * 3072);
  cvt(hidden, hid_bf,   2048ull * 3072);
  cvt(enc,    enc_bf,    256ull * 3072);

  gemm_wmma_bf16<<<dim3(9216 / 128, 2048 / 128), 256, 0, stream>>>(
      hid_bf, wqkv_bf, b_qkv, qkv, 2048, 9216, 3072);
  gemm_wmma_bf16<<<dim3(9216 / 128, 256 / 128), 256, 0, stream>>>(
      enc_bf, waqkv_bf, b_aqkv, eqkv, 256, 9216, 3072);

  qkv_prep<<<2304, 128, 0, stream>>>(qkv, eqkv, cosb, sinb,
                                     nq, nk, naq, nak, Qb, Kb, Vt);

  attn_wmma<<<dim3(36, 24), 128, 0, stream>>>(Qb, Kb, Vt, attn);

  gemm_wmma_bf16<<<dim3(3072 / 128, 2048 / 128), 256, 0, stream>>>(
      attn, wout_bf, b_out, out, 2048, 3072, 3072);
  gemm_wmma_bf16<<<dim3(3072 / 128, 256 / 128), 256, 0, stream>>>(
      attn + 2048ull * 3072, waout_bf, b_aout, out + 2048ull * 3072,
      256, 3072, 3072);
}
